// TemporalSceneGraphModel_13606456394107
// MI455X (gfx1250) — matrix operand order repaired and checked by reading, not compiled, not run on hardware
//
#include <hip/hip_runtime.h>
#include <hip/hip_bf16.h>

#define D_DIM   256
#define NGATE   768
#define NN      32          // nodes per frame
#define NE      992         // edges per frame
#define NBS     128         // frames
#define M_NODE  (NBS * NN)  // 4096 node rows
#define M_EDGE  (NBS * NE)  // 126976 edge rows

typedef __attribute__((ext_vector_type(16))) __bf16 v16bf;
typedef __attribute__((ext_vector_type(8)))  __bf16 v8bf;
typedef __attribute__((ext_vector_type(8)))  float  v8f;

static __device__ __forceinline__ __bf16 f2bf(float f) {
  unsigned u = __builtin_bit_cast(unsigned, f);
  u += 0x7fffu + ((u >> 16) & 1u);                // round-to-nearest-even
  unsigned short s = (unsigned short)(u >> 16);
  return __builtin_bit_cast(__bf16, s);
}

static __device__ __forceinline__ float sigf(float x) {
  return 1.0f / (1.0f + __expf(-x));
}

// ---------------------------------------------------------------- f32 -> bf16
__global__ void k_cvt_bf16(const float* __restrict__ src,
                           __bf16* __restrict__ dst, int n) {
  int i = blockIdx.x * blockDim.x + threadIdx.x;
  if (i < n) dst[i] = f2bf(src[i]);
}

// ------------------------------------------------------------- fused GRU cell
// One wave per 16-row tile; 8 waves (128 rows) per block.
// h' = (1-z)*tanh(in + r*hn) + z*h with r,z from sigmoid; all GEMM work done
// with v_wmma_f32_16x16x32_bf16, gate math fused in the epilogue.
// Templated on INIT (h == 0) so each specialization is branch-free: the init
// version has only the 3 x-side WMMAs, the steady-state version a straight
// 6-WMMA loop body.
template <bool INIT>
__global__ void __launch_bounds__(256) k_gru(
    const __bf16* __restrict__ xbf,   // [M,256] bf16 input message
    const __bf16* __restrict__ wih,   // [768,256] bf16
    const __bf16* __restrict__ whh,   // [768,256] bf16
    const float*  __restrict__ bih,   // [768]
    const float*  __restrict__ bhh,   // [768]
    float*        __restrict__ h32,   // [M,256] f32 master hidden (in/out)
    __bf16*       __restrict__ hbf)   // [M,256] bf16 mirror      (in/out)
{
  const int lane = threadIdx.x & 31;
  const int wav  = threadIdx.x >> 5;
  const int m0   = (blockIdx.x * 8 + wav) * 16;
  const int row  = m0 + (lane & 15);
  const int sel  = lane >> 4;     // half-wave select
  const int ka   = sel * 16;      // A frag: lanes 0-15 K 0-15, 16-31 K 16-31
  const int kb   = sel * 16;      // B frag: lanes 0-15 K 0-15, 16-31 K 16-31

  // Preload full-K A fragments (x and h). Preloading h also makes the
  // in-place hidden update safe: all h reads finish before any writes.
  v16bf ax[8], ah[8];
  {
    const __bf16* xr = xbf + (size_t)row * D_DIM;
    #pragma unroll
    for (int k = 0; k < 8; ++k) {
      v8bf lo = *(const v8bf*)(xr + k * 32 + ka);
      v8bf hi = *(const v8bf*)(xr + k * 32 + ka + 8);
      #pragma unroll
      for (int i = 0; i < 8; ++i) { ax[k][i] = lo[i]; ax[k][8 + i] = hi[i]; }
    }
  }
  if constexpr (!INIT) {
    const __bf16* hr = hbf + (size_t)row * D_DIM;
    #pragma unroll
    for (int k = 0; k < 8; ++k) {
      v8bf lo = *(const v8bf*)(hr + k * 32 + ka);
      v8bf hi = *(const v8bf*)(hr + k * 32 + ka + 8);
      #pragma unroll
      for (int i = 0; i < 8; ++i) { ah[k][i] = lo[i]; ah[k][8 + i] = hi[i]; }
    }
  }

  for (int nt = 0; nt < 16; ++nt) {
    const int d0 = nt * 16;
    const int nr = d0 + (lane & 15);   // weight row: r gate; +256 z; +512 n

    const __bf16* wi_r = wih + (size_t)nr * D_DIM + kb;
    const __bf16* wi_z = wih + (size_t)(nr + 256) * D_DIM + kb;
    const __bf16* wi_n = wih + (size_t)(nr + 512) * D_DIM + kb;
    const __bf16* wh_r = whh + (size_t)nr * D_DIM + kb;
    const __bf16* wh_z = whh + (size_t)(nr + 256) * D_DIM + kb;
    const __bf16* wh_n = whh + (size_t)(nr + 512) * D_DIM + kb;

    if (nt + 1 < 16) {                       // warm next tile's weight rows
      __builtin_prefetch(wi_r + 16 * D_DIM, 0, 1);
      if constexpr (!INIT) __builtin_prefetch(wh_r + 16 * D_DIM, 0, 1);
    }

    v8f air = {}, aiz = {}, ain = {}, ahr = {}, ahz = {}, ahn = {};

    #pragma unroll
    for (int k = 0; k < 8; ++k) {
      v16bf br = *(const v16bf*)(wi_r + k * 32);
      v16bf bz = *(const v16bf*)(wi_z + k * 32);
      v16bf bn = *(const v16bf*)(wi_n + k * 32);
      air = __builtin_amdgcn_wmma_f32_16x16x32_bf16(false, ax[k], false, br, (short)0, air, false, false);
      aiz = __builtin_amdgcn_wmma_f32_16x16x32_bf16(false, ax[k], false, bz, (short)0, aiz, false, false);
      ain = __builtin_amdgcn_wmma_f32_16x16x32_bf16(false, ax[k], false, bn, (short)0, ain, false, false);
      if constexpr (!INIT) {
        v16bf cr = *(const v16bf*)(wh_r + k * 32);
        v16bf cz = *(const v16bf*)(wh_z + k * 32);
        v16bf cn = *(const v16bf*)(wh_n + k * 32);
        ahr = __builtin_amdgcn_wmma_f32_16x16x32_bf16(false, ah[k], false, cr, (short)0, ahr, false, false);
        ahz = __builtin_amdgcn_wmma_f32_16x16x32_bf16(false, ah[k], false, cz, (short)0, ahz, false, false);
        ahn = __builtin_amdgcn_wmma_f32_16x16x32_bf16(false, ah[k], false, cn, (short)0, ahn, false, false);
      }
    }

    const float bir = bih[nr], biz = bih[nr + 256], bin_ = bih[nr + 512];
    const float bhr = bhh[nr], bhz = bhh[nr + 256], bhn = bhh[nr + 512];

    // C/D layout: VGPR j, lane L -> (m0 + j + 8*sel, d0 + L%16)
    #pragma unroll
    for (int j = 0; j < 8; ++j) {
      const int m = m0 + sel * 8 + j;
      const size_t o = (size_t)m * D_DIM + d0 + (lane & 15);
      const float hold = INIT ? 0.0f : h32[o];
      const float r  = sigf(air[j] + bir + ahr[j] + bhr);
      const float z  = sigf(aiz[j] + biz + ahz[j] + bhz);
      const float nn = tanhf(ain[j] + bin_ + r * (ahn[j] + bhn));
      const float hv = (1.0f - z) * nn + z * hold;
      h32[o] = hv;
      hbf[o] = f2bf(hv);
    }
  }
}

// -------------------------------------------- per-edge gates + edge messages
// One wave per edge; lane handles 8 contiguous dims; wave32 shfl reduction.
__global__ void __launch_bounds__(256) k_edge_gates(
    const float* __restrict__ node_h,   // [BS*32,256]
    const float* __restrict__ edge_h,   // [BS*992,256]
    const float* __restrict__ wnp,      // [512]
    const float* __restrict__ wes,      // [512]
    const float* __restrict__ weo,      // [512]
    const int*   __restrict__ e2n,      // [BS*992*3]
    float*       __restrict__ gate_n,   // [BS*992]  sigmoid(cat_s . w_node_pool)
    __bf16*      __restrict__ emsg)     // [BS*992,256] bf16 edge message
{
  const int lane = threadIdx.x & 31;
  const int wav  = threadIdx.x >> 5;
  const int ge   = blockIdx.x * 8 + wav;
  const int b    = ge / NE;
  const int sub  = e2n[(size_t)ge * 3 + 1];
  const int obj  = e2n[(size_t)ge * 3 + 2];
  const float* ns = node_h + (size_t)(b * NN + sub) * D_DIM;
  const float* no = node_h + (size_t)(b * NN + obj) * D_DIM;
  const float* eh = edge_h + (size_t)ge * D_DIM;
  const int d = lane * 8;

  float sn = 0.f, ss = 0.f, so = 0.f;
  float sv[8], ov[8];
  #pragma unroll
  for (int i = 0; i < 8; ++i) {
    const float s = ns[d + i], o = no[d + i], e = eh[d + i];
    sv[i] = s; ov[i] = o;
    sn += s * wnp[d + i] + e * wnp[256 + d + i];
    ss += s * wes[d + i] + e * wes[256 + d + i];
    so += o * weo[d + i] + e * weo[256 + d + i];
  }
  #pragma unroll
  for (int m = 16; m >= 1; m >>= 1) {
    sn += __shfl_xor(sn, m, 32);
    ss += __shfl_xor(ss, m, 32);
    so += __shfl_xor(so, m, 32);
  }
  const float gs = sigf(ss), go = sigf(so);
  #pragma unroll
  for (int i = 0; i < 8; ++i)
    emsg[(size_t)ge * D_DIM + d + i] = f2bf(gs * sv[i] + go * ov[i]);
  if (lane == 0) gate_n[ge] = sigf(sn);
}

// ------------------------------------- node messages via closed-form gather
// edge(i,j) index = i*31 + (j - (j>i)); node n gets 31 subject edges
// (contiguous block n*31..n*31+30) plus 31 object edges. No atomics needed.
__global__ void __launch_bounds__(256) k_node_msg(
    const float* __restrict__ edge_h,
    const float* __restrict__ gate_n,
    __bf16*      __restrict__ nmsg)     // [BS*32,256] bf16
{
  const int b = blockIdx.x >> 5;
  const int n = blockIdx.x & 31;
  __shared__ float g_sh[62];
  __shared__ int   e_sh[62];
  const int t = threadIdx.x;
  if (t < 62) {
    int e;
    if (t < 31) {
      e = n * 31 + t;                         // (n, j) subject side
    } else {
      const int t2 = t - 31;
      const int i  = (t2 < n) ? t2 : t2 + 1;  // (i, n) object side
      e = i * 31 + ((i < n) ? (n - 1) : n);
    }
    e_sh[t] = e;
    g_sh[t] = gate_n[(size_t)b * NE + e];
  }
  __syncthreads();
  const float* ehb = edge_h + (size_t)b * NE * D_DIM;
  float acc = 0.0f;
  #pragma unroll 2
  for (int k = 0; k < 62; ++k)
    acc += g_sh[k] * ehb[(size_t)e_sh[k] * D_DIM + t];
  nmsg[(size_t)(b * NN + n) * D_DIM + t] = f2bf(acc);
}

// ---------------------------------------------------------------------- host
extern "C" void kernel_launch(void* const* d_in, const int* in_sizes, int n_in,
                              void* d_out, int out_size, void* d_ws, size_t ws_size,
                              hipStream_t stream) {
  (void)in_sizes; (void)n_in; (void)out_size; (void)ws_size;

  const float* node_lat = (const float*)d_in[0];
  const float* edge_lat = (const float*)d_in[1];
  const float* nWih = (const float*)d_in[2];
  const float* nWhh = (const float*)d_in[3];
  const float* nbih = (const float*)d_in[4];
  const float* nbhh = (const float*)d_in[5];
  const float* eWih = (const float*)d_in[6];
  const float* eWhh = (const float*)d_in[7];
  const float* ebih = (const float*)d_in[8];
  const float* ebhh = (const float*)d_in[9];
  const float* wnp  = (const float*)d_in[10];
  const float* wes  = (const float*)d_in[11];
  const float* weo  = (const float*)d_in[12];
  const int*   e2n  = (const int*)d_in[13];

  // Outputs double as the live f32 hidden state (node block then edge block).
  float* node_h = (float*)d_out;
  float* edge_h = node_h + (size_t)M_NODE * D_DIM;

  char* ws = (char*)d_ws;
  size_t off = 0;
  auto take = [&](size_t bytes) -> void* {
    void* p = ws + off;
    off = (off + bytes + 255) & ~(size_t)255;
    return p;
  };
  __bf16* nWih_bf = (__bf16*)take((size_t)NGATE * D_DIM * 2);
  __bf16* nWhh_bf = (__bf16*)take((size_t)NGATE * D_DIM * 2);
  __bf16* eWih_bf = (__bf16*)take((size_t)NGATE * D_DIM * 2);
  __bf16* eWhh_bf = (__bf16*)take((size_t)NGATE * D_DIM * 2);
  __bf16* node_x  = (__bf16*)take((size_t)M_NODE * D_DIM * 2);  // x / node msg
  __bf16* edge_x  = (__bf16*)take((size_t)M_EDGE * D_DIM * 2);  // x / edge msg
  __bf16* node_hb = (__bf16*)take((size_t)M_NODE * D_DIM * 2);  // bf16 mirror
  __bf16* edge_hb = (__bf16*)take((size_t)M_EDGE * D_DIM * 2);  // bf16 mirror
  float*  gate_n  = (float*) take((size_t)M_EDGE * 4);

  auto cvt = [&](const float* s, __bf16* d, int n) {
    k_cvt_bf16<<<(n + 255) / 256, 256, 0, stream>>>(s, d, n);
  };
  cvt(nWih, nWih_bf, NGATE * D_DIM);
  cvt(nWhh, nWhh_bf, NGATE * D_DIM);
  cvt(eWih, eWih_bf, NGATE * D_DIM);
  cvt(eWhh, eWhh_bf, NGATE * D_DIM);
  cvt(node_lat, node_x, M_NODE * D_DIM);
  cvt(edge_lat, edge_x, M_EDGE * D_DIM);

  // Initial GRU with h = 0 (skips the h GEMM entirely).
  k_gru<true><<<M_NODE / 128, 256, 0, stream>>>(node_x, nWih_bf, nWhh_bf,
                                                nbih, nbhh, node_h, node_hb);
  k_gru<true><<<M_EDGE / 128, 256, 0, stream>>>(edge_x, eWih_bf, eWhh_bf,
                                                ebih, ebhh, edge_h, edge_hb);

  for (int it = 0; it < 3; ++it) {
    k_edge_gates<<<M_EDGE / 8, 256, 0, stream>>>(node_h, edge_h, wnp, wes, weo,
                                                 e2n, gate_n, edge_x);
    k_node_msg<<<NBS * NN, 256, 0, stream>>>(edge_h, gate_n, node_x);
    k_gru<false><<<M_NODE / 128, 256, 0, stream>>>(node_x, nWih_bf, nWhh_bf,
                                                   nbih, nbhh, node_h, node_hb);
    k_gru<false><<<M_EDGE / 128, 256, 0, stream>>>(edge_x, eWih_bf, eWhh_bf,
                                                   ebih, ebhh, edge_h, edge_hb);
  }
}
